// Decoder_36833639531196
// MI455X (gfx1250) — matrix operand order repaired and checked by reading, not compiled, not run on hardware
//
#include <hip/hip_runtime.h>
#include <hip/hip_bf16.h>

// ---------------------------------------------------------------------------
// Show-and-tell LSTM decoder for MI455X (gfx1250, wave32, WMMA).
// All heavy GEMMs run as fp16 x fp16 -> f32 via v_wmma_f32_16x16x32_f16.
// Weights (~55MB fp16) are L2-resident; B tiles staged in double-buffered LDS
// so each weight element crosses L2->WGP once per GEMM; A fragments are
// software-pipelined one 64-K chunk ahead (2x-unrolled ping-pong, no copies).
// sched_group_barrier shapes each chunk into VMEM -> DS-write -> DS-read -> WMMA.
// Batch M=128 == 8 waves x 16 rows.
// ---------------------------------------------------------------------------

typedef __attribute__((ext_vector_type(16))) _Float16 v16h;
typedef __attribute__((ext_vector_type(8)))  _Float16 v8h;
typedef __attribute__((ext_vector_type(8)))  float    v8f;

#define BB     128
#define TT     20
#define NN     49
#define VOCAB  10000
#define EMB    512
#define HDIM   1024
#define ATT    512
#define VDIMM  512
#define LDP    72        // LDS row pitch in halves (144B -> conflict-free, 16B aligned)

// Shape one K-chunk's scheduling region: batch loads, then back-to-back WMMAs.
// Masks (LLVM AMDGPU): 0x020 VMEM_READ, 0x200 DS_WRITE, 0x100 DS_READ, 0x008 MFMA/WMMA.
#if __has_builtin(__builtin_amdgcn_sched_group_barrier)
#define SCHED_PIPE() do {                                   \
    __builtin_amdgcn_sched_group_barrier(0x020, 6, 0);      \
    __builtin_amdgcn_sched_group_barrier(0x200, 2, 0);      \
    __builtin_amdgcn_sched_group_barrier(0x100, 16, 0);     \
    __builtin_amdgcn_sched_group_barrier(0x008, 8, 0);      \
} while (0)
#else
#define SCHED_PIPE() do {} while (0)
#endif

// ---------------------------------------------------------------------------
// Generic WMMA GEMM: C[M,N] = act( A[M,K] * Bm[N,K]^T + bias )
//  - A, Bm fp16 row-major; K % 128 == 0; M = gridDim.y * 128
//  - block = 256 thr (8 waves); wave w owns rows [mb + 16w, mb + 16w + 16)
//  - block computes 64 N-columns; B chunks (64 x 64 halves) double-buffered in LDS
//  - A fragments prefetched one chunk ahead into ping-pong registers
//  - all 8 B fragments of a chunk loaded before the 8 back-to-back WMMAs
//  - optional fp32 store (Cf, ldc) and fp16 store (Ch, ldch); act: 0=id, 1=tanh
// ---------------------------------------------------------------------------
__global__ __launch_bounds__(256, 1) void gemm_wmma_f16(
    const _Float16* __restrict__ A, const _Float16* __restrict__ Bm,
    int K, int N, const float* __restrict__ bias, int act,
    float* __restrict__ Cf, int ldc, _Float16* __restrict__ Ch, int ldch)
{
    __shared__ _Float16 bl[2][64 * LDP];

    const int  tid  = threadIdx.x;
    const int  lane = tid & 31;
    const int  wid  = tid >> 5;          // 0..7 : m sub-tile within the 128-row block
    const int  l    = lane & 15;
    const int  hseg = lane >> 4;         // 0/1  : K half-segment per ISA fragment layout
    const int  nb   = blockIdx.x * 64;
    const long mb   = (long)blockIdx.y * 128;
    const long arow = mb + wid * 16 + l;

    v8f acc[4] = {};                     // 4 tiles along N => 16(M) x 64(N) per wave

    // B-chunk fill assignment: 256 thr -> 64 rows x 4 segments of 16 halves
    const int fr = tid >> 2;
    const int fs = tid & 3;
    int br = nb + fr; if (br >= N) br = N - 1;          // clamp for N=10000 edge
    const _Float16* bsrc = Bm + (size_t)br * K + fs * 16;

    // global -> LDS stage of one 64x64-half B chunk (one v16h load / thread)
    auto fillB = [&](int buf, int k0) {
        union { v16h v; v8h h[2]; } bu;
        bu.v = *(const v16h*)(bsrc + k0);
        _Float16* d = &bl[buf][fr * LDP + fs * 16];
        *(v8h*)d       = bu.h[0];
        *(v8h*)(d + 8) = bu.h[1];
    };
    // A fragments (16x32 f16 each) for one 64-K chunk:
    // lanes 0-15 -> K {0..7,16..23}; lanes 16-31 -> K {8..15,24..31}
    auto loadA = [&](int k0, v16h& x0, v16h& x1) {
        const _Float16* ap = A + arow * (long)K + k0 + hseg * 8;
        union { v16h v; v8h h[2]; } u0, u1;
        u0.h[0] = *(const v8h*)ap;        u0.h[1] = *(const v8h*)(ap + 16);
        u1.h[0] = *(const v8h*)(ap + 32); u1.h[1] = *(const v8h*)(ap + 48);
        x0 = u0.v; x1 = u1.v;
    };
    // one 64-K chunk of matrix math: load all 8 B fragments, then 8 WMMAs
    auto compute = [&](int buf, v16h x0, v16h x1) {
        const _Float16* lb = &bl[buf][(size_t)l * LDP + hseg * 8];
        union { v16h v; v8h h[2]; } b[8];
        #pragma unroll
        for (int i = 0; i < 4; ++i) {
            const _Float16* p = lb + i * 16 * LDP;
            b[i    ].h[0] = *(const v8h*)p;          b[i    ].h[1] = *(const v8h*)(p + 16);
            b[i + 4].h[0] = *(const v8h*)(p + 32);   b[i + 4].h[1] = *(const v8h*)(p + 48);
        }
        #pragma unroll
        for (int i = 0; i < 4; ++i)
            acc[i] = __builtin_amdgcn_wmma_f32_16x16x32_f16(
                false, x0, false, b[i].v, (short)0, acc[i], false, false);
        #pragma unroll
        for (int i = 0; i < 4; ++i)
            acc[i] = __builtin_amdgcn_wmma_f32_16x16x32_f16(
                false, x1, false, b[i + 4].v, (short)0, acc[i], false, false);
    };

    v16h aE0, aE1, aO0, aO1;             // even/odd chunk A ping-pong registers
    loadA(0, aE0, aE1);
    fillB(0, 0);
    __syncthreads();

    const int nchunks = K >> 6;          // even (K % 128 == 0)
    for (int ci = 0; ci < nchunks; ci += 2) {
        // even chunk ci: compute bl[0]; stage chunk ci+1 -> bl[1], A(ci+1) -> odd regs
        fillB(1, (ci + 1) << 6);
        loadA((ci + 1) << 6, aO0, aO1);
        if (((ci + 2) << 6) < K) __builtin_prefetch(bsrc + ((ci + 2) << 6), 0, 1);
        compute(0, aE0, aE1);
        SCHED_PIPE();
        __syncthreads();
        // odd chunk ci+1: compute bl[1]; stage chunk ci+2 -> bl[0], A(ci+2) -> even regs
        if (ci + 2 < nchunks) {
            fillB(0, (ci + 2) << 6);
            loadA((ci + 2) << 6, aE0, aE1);
        }
        compute(1, aO0, aO1);
        SCHED_PIPE();
        __syncthreads();
    }

    // Epilogue. C 16x16 f32 layout: VGPR v, lanes 0-15 -> M=v, lanes 16-31 -> M=8+v.
    #pragma unroll
    for (int nt = 0; nt < 4; ++nt) {
        int col = nb + nt * 16 + l;
        if (col >= N) continue;
        float bv = bias ? bias[col] : 0.0f;
        #pragma unroll
        for (int v = 0; v < 8; ++v) {
            long row = mb + wid * 16 + hseg * 8 + v;
            float r = acc[nt][v] + bv;
            if (act == 1) r = tanhf(r);
            if (Cf) Cf[row * (long)ldc + col] = r;
            if (Ch) Ch[row * (long)ldch + col] = (_Float16)r;
        }
    }
}

// ---------------------------------------------------------------------------
// Fused LSTM cell + LayerNorm. One block per batch row. Gate order i,f,g,o.
// Writes normalized h (fp16) into two destinations (next-GEMM A buffers).
// ---------------------------------------------------------------------------
__global__ __launch_bounds__(256) void lstm_cell_ln(
    const float* __restrict__ gates, const float* __restrict__ bih,
    const float* __restrict__ bhh, float* __restrict__ c,
    const float* __restrict__ g_ln, const float* __restrict__ b_ln,
    _Float16* __restrict__ o1, int ld1, _Float16* __restrict__ o2, int ld2)
{
    __shared__ float hb[HDIM];
    __shared__ float red[16];
    const int b = blockIdx.x, tid = threadIdx.x;
    const float* gb = gates + (size_t)b * 4 * HDIM;

    for (int h = tid; h < HDIM; h += 256) {
        float ig = gb[h]            + bih[h]            + bhh[h];
        float fg = gb[HDIM + h]     + bih[HDIM + h]     + bhh[HDIM + h];
        float gg = gb[2 * HDIM + h] + bih[2 * HDIM + h] + bhh[2 * HDIM + h];
        float og = gb[3 * HDIM + h] + bih[3 * HDIM + h] + bhh[3 * HDIM + h];
        float si = 1.0f / (1.0f + __expf(-ig));
        float sf = 1.0f / (1.0f + __expf(-fg));
        float so = 1.0f / (1.0f + __expf(-og));
        float cn = sf * c[(size_t)b * HDIM + h] + si * tanhf(gg);
        c[(size_t)b * HDIM + h] = cn;
        hb[h] = so * tanhf(cn);
    }
    __syncthreads();

    float s = 0.0f, s2 = 0.0f;
    for (int h = tid; h < HDIM; h += 256) { float v = hb[h]; s += v; s2 += v * v; }
    for (int off = 16; off > 0; off >>= 1) { s += __shfl_down(s, off); s2 += __shfl_down(s2, off); }
    if ((tid & 31) == 0) { red[tid >> 5] = s; red[8 + (tid >> 5)] = s2; }
    __syncthreads();
    if (tid == 0) {
        float S = 0.0f, S2 = 0.0f;
        for (int w = 0; w < 8; ++w) { S += red[w]; S2 += red[8 + w]; }
        red[0] = S; red[8] = S2;
    }
    __syncthreads();
    float mu  = red[0] * (1.0f / HDIM);
    float var = red[8] * (1.0f / HDIM) - mu * mu;
    float inv = rsqrtf(var + 1e-5f);
    for (int h = tid; h < HDIM; h += 256) {
        float v = (hb[h] - mu) * inv * g_ln[h] + b_ln[h];
        o1[(size_t)b * ld1 + h] = (_Float16)v;
        o2[(size_t)b * ld2 + h] = (_Float16)v;
    }
}

// ---------------------------------------------------------------------------
// Fused attention (e = tanh(q + Uv) . attv -> softmax(N=49) -> ctx = a . Vp)
// plus embedding gather for timestep t. One block per batch row.
// Writes xcat1[:, 0:512] = emb(y[:,t]), xcat1[:, 512:1536] = ctx (fp16).
// ---------------------------------------------------------------------------
__global__ __launch_bounds__(256) void attention_ctx_emb(
    const float* __restrict__ q, const float* __restrict__ Uv,
    const float* __restrict__ attv, const float* __restrict__ Vp,
    const float* __restrict__ embedW, const int* __restrict__ y, int t,
    _Float16* __restrict__ xcat1)
{
    __shared__ float e_sh[64];
    const int b = blockIdx.x, tid = threadIdx.x, lane = tid & 31, wv = tid >> 5;
    const float* qb = q + (size_t)b * ATT;

    for (int n = wv; n < NN; n += 8) {
        const float* uv = Uv + ((size_t)b * NN + n) * ATT;
        float s = 0.0f;
        for (int k = lane; k < ATT; k += 32) s += tanhf(qb[k] + uv[k]) * attv[k];
        for (int off = 16; off > 0; off >>= 1) s += __shfl_down(s, off);
        if (lane == 0) e_sh[n] = s;
    }
    __syncthreads();
    if (tid == 0) {
        float m = -1e30f;
        for (int n = 0; n < NN; ++n) m = fmaxf(m, e_sh[n]);
        float s = 0.0f;
        for (int n = 0; n < NN; ++n) { float e = __expf(e_sh[n] - m); e_sh[n] = e; s += e; }
        float inv = 1.0f / s;
        for (int n = 0; n < NN; ++n) e_sh[n] *= inv;
    }
    __syncthreads();

    _Float16* xb = xcat1 + (size_t)b * 2560;
    for (int h = tid; h < HDIM; h += 256) {
        float acc = 0.0f;
        for (int n = 0; n < NN; ++n) acc += e_sh[n] * Vp[((size_t)b * NN + n) * HDIM + h];
        xb[EMB + h] = (_Float16)acc;
    }
    int tok = y[b * TT + t];
    for (int e = tid; e < EMB; e += 256) xb[e] = (_Float16)embedW[(size_t)tok * EMB + e];
}

// ---------------------------------------------------------------------------
// feat_mean = Vp.mean(axis=1) -> fp16  (one block per batch row)
// ---------------------------------------------------------------------------
__global__ __launch_bounds__(256) void mean_feat(const float* __restrict__ Vp,
                                                 _Float16* __restrict__ mh)
{
    const int b = blockIdx.x, tid = threadIdx.x;
    for (int h = tid; h < HDIM; h += 256) {
        float s = 0.0f;
        for (int n = 0; n < NN; ++n) s += Vp[((size_t)b * NN + n) * HDIM + h];
        mh[(size_t)b * HDIM + h] = (_Float16)(s * (1.0f / NN));
    }
}

// ---------------------------------------------------------------------------
// Utility kernels (weight prep / zero-init) — all grid-stride, deterministic.
// ---------------------------------------------------------------------------
__global__ void cvt_f32_f16(const float* __restrict__ src, _Float16* __restrict__ dst, long n)
{
    long i = (long)blockIdx.x * blockDim.x + threadIdx.x;
    long stride = (long)gridDim.x * blockDim.x;
    for (; i < n; i += stride) dst[i] = (_Float16)src[i];
}

__global__ void concat_cvt_f16(const float* __restrict__ A, int Ka,
                               const float* __restrict__ Bs, int Kb,
                               _Float16* __restrict__ dst, long rows)
{
    long n = rows * (long)(Ka + Kb);
    long i = (long)blockIdx.x * blockDim.x + threadIdx.x;
    long stride = (long)gridDim.x * blockDim.x;
    for (; i < n; i += stride) {
        long r = i / (Ka + Kb);
        int  c = (int)(i - r * (Ka + Kb));
        float v = (c < Ka) ? A[r * Ka + c] : Bs[r * Kb + (c - Ka)];
        dst[i] = (_Float16)v;
    }
}

__global__ void zero_u32(unsigned int* __restrict__ p, long n)
{
    long i = (long)blockIdx.x * blockDim.x + threadIdx.x;
    long stride = (long)gridDim.x * blockDim.x;
    for (; i < n; i += stride) p[i] = 0u;
}

// ---------------------------------------------------------------------------
extern "C" void kernel_launch(void* const* d_in, const int* in_sizes, int n_in,
                              void* d_out, int out_size, void* d_ws, size_t ws_size,
                              hipStream_t stream)
{
    (void)in_sizes; (void)n_in; (void)out_size; (void)ws_size;

    const float* V       = (const float*)d_in[0];
    const int*   y       = (const int*)  d_in[1];
    const float* embed_W = (const float*)d_in[2];
    const float* Vp_W    = (const float*)d_in[3];
    const float* Vp_b    = (const float*)d_in[4];
    const float* attW    = (const float*)d_in[5];
    const float* attU    = (const float*)d_in[6];
    const float* attv    = (const float*)d_in[7];
    const float* l1_Wih  = (const float*)d_in[8];
    const float* l1_Whh  = (const float*)d_in[9];
    const float* l1_bih  = (const float*)d_in[10];
    const float* l1_bhh  = (const float*)d_in[11];
    const float* l2_Wih  = (const float*)d_in[12];
    const float* l2_Whh  = (const float*)d_in[13];
    const float* l2_bih  = (const float*)d_in[14];
    const float* l2_bhh  = (const float*)d_in[15];
    const float* n1_g    = (const float*)d_in[16];
    const float* n1_b    = (const float*)d_in[17];
    const float* n2_g    = (const float*)d_in[18];
    const float* n2_b    = (const float*)d_in[19];
    const float* ih_W    = (const float*)d_in[20];
    const float* ih_b    = (const float*)d_in[21];
    const float* ic_W    = (const float*)d_in[22];
    const float* ic_b    = (const float*)d_in[23];
    const float* proj_W  = (const float*)d_in[24];
    float* out = (float*)d_out;                  // [B, T-1, VOCAB] fp32

    // ---- workspace carve-out (256B aligned) ----
    char* wp = (char*)d_ws;
    auto alloc = [&](size_t bytes) -> void* {
        void* r = (void*)wp;
        wp += (bytes + 255) & ~(size_t)255;
        return r;
    };
    const long MV = (long)BB * NN;               // 6272
    _Float16* Vh     = (_Float16*)alloc((size_t)MV * VDIMM * 2);
    _Float16* VpWh   = (_Float16*)alloc((size_t)HDIM * VDIMM * 2);
    _Float16* attUh  = (_Float16*)alloc((size_t)ATT * HDIM * 2);
    _Float16* attWh  = (_Float16*)alloc((size_t)ATT * HDIM * 2);
    _Float16* ihWh   = (_Float16*)alloc((size_t)HDIM * HDIM * 2);
    _Float16* icWh   = (_Float16*)alloc((size_t)HDIM * HDIM * 2);
    _Float16* W1cat  = (_Float16*)alloc((size_t)4 * HDIM * 2560 * 2);
    _Float16* W2cat  = (_Float16*)alloc((size_t)4 * HDIM * 2048 * 2);
    _Float16* projWh = (_Float16*)alloc((size_t)EMB * HDIM * 2);
    _Float16* embedh = (_Float16*)alloc((size_t)VOCAB * EMB * 2);
    float*    Vp     = (float*)   alloc((size_t)MV * HDIM * 4);
    _Float16* Vph    = (_Float16*)alloc((size_t)MV * HDIM * 2);
    float*    Uv     = (float*)   alloc((size_t)MV * ATT * 4);
    _Float16* meanh  = (_Float16*)alloc((size_t)BB * HDIM * 2);
    _Float16* xcat1  = (_Float16*)alloc((size_t)BB * 2560 * 2);   // [emb|ctx|h1]
    _Float16* xcat2  = (_Float16*)alloc((size_t)BB * 2048 * 2);   // [h1n|h2]
    float*    c1     = (float*)   alloc((size_t)BB * HDIM * 4);
    float*    c2     = (float*)   alloc((size_t)BB * HDIM * 4);
    _Float16* h2h    = (_Float16*)alloc((size_t)BB * HDIM * 2);
    float*    qbuf   = (float*)   alloc((size_t)BB * ATT * 4);
    float*    gates  = (float*)   alloc((size_t)BB * 4 * HDIM * 4);
    _Float16* ph     = (_Float16*)alloc((size_t)BB * EMB * 2);

    const dim3 blk(256);
    const dim3 gcv(512);

    // ---- phase 0: weight prep (fp32 -> fp16, concatenated LSTM weights) ----
    cvt_f32_f16<<<gcv, blk, 0, stream>>>(V,       Vh,     MV * (long)VDIMM);
    cvt_f32_f16<<<gcv, blk, 0, stream>>>(Vp_W,    VpWh,   (long)HDIM * VDIMM);
    cvt_f32_f16<<<gcv, blk, 0, stream>>>(attU,    attUh,  (long)ATT * HDIM);
    cvt_f32_f16<<<gcv, blk, 0, stream>>>(attW,    attWh,  (long)ATT * HDIM);
    cvt_f32_f16<<<gcv, blk, 0, stream>>>(ih_W,    ihWh,   (long)HDIM * HDIM);
    cvt_f32_f16<<<gcv, blk, 0, stream>>>(ic_W,    icWh,   (long)HDIM * HDIM);
    cvt_f32_f16<<<gcv, blk, 0, stream>>>(proj_W,  projWh, (long)EMB * HDIM);
    cvt_f32_f16<<<gcv, blk, 0, stream>>>(embed_W, embedh, (long)VOCAB * EMB);
    concat_cvt_f16<<<gcv, blk, 0, stream>>>(l1_Wih, EMB + HDIM, l1_Whh, HDIM, W1cat, 4 * HDIM);
    concat_cvt_f16<<<gcv, blk, 0, stream>>>(l2_Wih, HDIM,       l2_Whh, HDIM, W2cat, 4 * HDIM);
    zero_u32<<<gcv, blk, 0, stream>>>((unsigned int*)c2,    (long)BB * HDIM);
    zero_u32<<<gcv, blk, 0, stream>>>((unsigned int*)h2h,   (long)BB * HDIM / 2);
    zero_u32<<<gcv, blk, 0, stream>>>((unsigned int*)xcat2, (long)BB * 2048 / 2);

    // ---- phase 1: Vp, feat_mean, h1/c1 init, Uv ----
    gemm_wmma_f16<<<dim3(HDIM / 64, MV / 128), blk, 0, stream>>>(
        Vh, VpWh, VDIMM, HDIM, Vp_b, 0, Vp, HDIM, Vph, HDIM);
    mean_feat<<<BB, blk, 0, stream>>>(Vp, meanh);
    gemm_wmma_f16<<<dim3(HDIM / 64, 1), blk, 0, stream>>>(      // h1 = tanh(mean@ihW^T+b)
        meanh, ihWh, HDIM, HDIM, ih_b, 1, nullptr, 0, xcat1 + 1536, 2560);
    gemm_wmma_f16<<<dim3(HDIM / 64, 1), blk, 0, stream>>>(      // c1 = tanh(mean@icW^T+b)
        meanh, icWh, HDIM, HDIM, ic_b, 1, c1, HDIM, nullptr, 0);
    gemm_wmma_f16<<<dim3(ATT / 64, MV / 128), blk, 0, stream>>>( // Uv = Vp@attU^T
        Vph, attUh, HDIM, ATT, nullptr, 0, Uv, ATT, nullptr, 0);

    // ---- phase 2: 19 sequential decode steps ----
    for (int t = 0; t < TT - 1; ++t) {
        gemm_wmma_f16<<<dim3(ATT / 64, 1), blk, 0, stream>>>(    // q = h2@attW^T
            h2h, attWh, HDIM, ATT, nullptr, 0, qbuf, ATT, nullptr, 0);
        attention_ctx_emb<<<BB, blk, 0, stream>>>(qbuf, Uv, attv, Vp, embed_W, y, t, xcat1);
        gemm_wmma_f16<<<dim3(4 * HDIM / 64, 1), blk, 0, stream>>>( // gates1
            xcat1, W1cat, 2560, 4 * HDIM, nullptr, 0, gates, 4 * HDIM, nullptr, 0);
        lstm_cell_ln<<<BB, blk, 0, stream>>>(gates, l1_bih, l1_bhh, c1, n1_g, n1_b,
                                             xcat2, 2048, xcat1 + 1536, 2560);
        gemm_wmma_f16<<<dim3(4 * HDIM / 64, 1), blk, 0, stream>>>( // gates2
            xcat2, W2cat, 2048, 4 * HDIM, nullptr, 0, gates, 4 * HDIM, nullptr, 0);
        lstm_cell_ln<<<BB, blk, 0, stream>>>(gates, l2_bih, l2_bhh, c2, n2_g, n2_b,
                                             h2h, HDIM, xcat2 + HDIM, 2048);
        gemm_wmma_f16<<<dim3(EMB / 64, 1), blk, 0, stream>>>(      // p = h2n@projW^T
            h2h, projWh, HDIM, EMB, nullptr, 0, nullptr, 0, ph, EMB);
        gemm_wmma_f16<<<dim3((VOCAB + 63) / 64, 1), blk, 0, stream>>>( // logits = p@embed^T
            ph, embedh, EMB, VOCAB, nullptr, 0,
            out + (size_t)t * VOCAB, (TT - 1) * VOCAB, nullptr, 0);
    }
}
